// deform_fusion_35570919146218
// MI455X (gfx1250) — matrix operand order repaired and checked by reading, not compiled, not run on hardware
//
#include <hip/hip_runtime.h>
#include <hip/hip_bf16.h>
#include <stdint.h>

// ---------------------------------------------------------------------------
//   cnn_feat: (8, 768, 32, 32)   vit_feat: (8, 3840, 16, 16)
//   w_off: (18, 3840, 3, 3)      w_def: (768, 768, 3, 3)
//   w1:    (512, 768, 3, 3)      w2:    (768, 512, 3, 3)
//   out:   (8, 768, 16, 16)
//
// GEMM operands live in global memory in *WMMA fragment order*:
//   A tile (16M x 32K bf16) = 256 u32:  [(mt*Kc + kc)*256 + lane*8 + j]
//   B tile (32K x 16N bf16) = 256 u32:  [(kc*Nt + nt)*256 + lane*8 + j]
// ---------------------------------------------------------------------------

typedef __attribute__((ext_vector_type(16))) __bf16       v16bf;
typedef __attribute__((ext_vector_type(8)))  float        v8f;
typedef __attribute__((ext_vector_type(8)))  unsigned int v8u;

union Frag { v8u u; v16bf b; };

__device__ __forceinline__ unsigned short f2bf(float f) {
  unsigned u = __float_as_uint(f);
  unsigned r = (u + 0x7FFFu + ((u >> 16) & 1u)) >> 16;   // RNE
  return (unsigned short)r;
}

__device__ __forceinline__ void async_b128(unsigned ldsAddr, const uint4* g) {
  unsigned long long ga = (unsigned long long)(size_t)g;
  asm volatile("global_load_async_to_lds_b128 %0, %1, off"
               :: "v"(ldsAddr), "v"(ga) : "memory");
}
__device__ __forceinline__ void wait_async_le(int) {}
__device__ __forceinline__ void wait_async_1() {
  asm volatile("s_wait_asynccnt 0x1" ::: "memory");
}
__device__ __forceinline__ void wait_async_0() {
  asm volatile("s_wait_asynccnt 0x0" ::: "memory");
}

// ---------------------------------------------------------------------------
// Weight repack fp32 -> bf16 fragment-order B (k = tap*C + c)
// ---------------------------------------------------------------------------
__global__ void k_pack_w(const float* __restrict__ w, unsigned* __restrict__ Bf,
                         int O, int C) {
  int Nt = O >> 4;
  size_t total = (size_t)9 * C * O / 2;
  size_t gs = (size_t)gridDim.x * blockDim.x;
  for (size_t i = (size_t)blockIdx.x * blockDim.x + threadIdx.x; i < total; i += gs) {
    int    j    = (int)(i & 7);
    int    lane = (int)((i >> 3) & 31);
    size_t tile = i >> 8;
    int    nt   = (int)(tile % Nt);
    int    kc   = (int)(tile / Nt);
    int half = lane >> 4;
    int n    = nt * 16 + (lane & 15);
    int k0   = kc * 32 + ((j >= 4) ? 16 : 0) + half * 8 + (j & 3) * 2;
    int tap  = k0 / C;
    int c    = k0 - tap * C;            // even; c+1 same tap (C even, chunk-aligned)
    unsigned lo = f2bf(w[((size_t)n * C + c) * 9 + tap]);
    unsigned hi = f2bf(w[((size_t)n * C + c + 1) * 9 + tap]);
    Bf[i] = lo | (hi << 16);
  }
}

// ---------------------------------------------------------------------------
// Offset conv (nearest upsample folded): fp32 VALU, N=18 too thin for WMMA.
// ---------------------------------------------------------------------------
__global__ void __launch_bounds__(512)
k_offset_conv(const float* __restrict__ vit, const float* __restrict__ w_off,
              const float* __restrict__ b_off, float* __restrict__ offt) {
  int b = blockIdx.x / 18;
  int o = blockIdx.x % 18;
  __shared__ float lv[8 * 256];
  int t = threadIdx.x;

  int   lidx[2][9];
  float msk[2][9];
#pragma unroll
  for (int i = 0; i < 2; ++i) {
    int p = t + i * 512;
    int y = p >> 5, x = p & 31;
#pragma unroll
    for (int tap = 0; tap < 9; ++tap) {
      int dy = tap / 3 - 1, dx = tap % 3 - 1;
      int yy = y + dy, xx = x + dx;
      bool ok = (yy >= 0) && (yy < 32) && (xx >= 0) && (xx < 32);
      lidx[i][tap] = (ok ? (yy >> 1) : 0) * 16 + (ok ? (xx >> 1) : 0);
      msk[i][tap]  = ok ? 1.0f : 0.0f;
    }
  }

  float acc0 = 0.0f, acc1 = 0.0f;
  const float* vb = vit + (size_t)b * 3840 * 256;
  const float* wb = w_off + (size_t)o * 3840 * 9;

  for (int c0 = 0; c0 < 3840; c0 += 8) {
    __syncthreads();
#pragma unroll
    for (int j = 0; j < 4; ++j) {
      int e = t + j * 512;
      lv[e] = vb[(size_t)c0 * 256 + e];
    }
    __syncthreads();
#pragma unroll 2
    for (int cc = 0; cc < 8; ++cc) {
      const float* lc = lv + cc * 256;
#pragma unroll
      for (int tap = 0; tap < 9; ++tap) {
        float wk = wb[(c0 + cc) * 9 + tap];
        acc0 = fmaf(lc[lidx[0][tap]] * msk[0][tap], wk, acc0);
        acc1 = fmaf(lc[lidx[1][tap]] * msk[1][tap], wk, acc1);
      }
    }
  }
  float bo = b_off[o];
  float* op = offt + (size_t)(b * 18 + o) * 1024;
  op[t]       = acc0 + bo;
  op[t + 512] = acc1 + bo;
}

// ---------------------------------------------------------------------------
// Deform bilinear sampling -> fragment-order A (Ktot = 6912, Kc = 216).
// ---------------------------------------------------------------------------
__global__ void __launch_bounds__(256)
k_deform_sample(const float* __restrict__ x, const float* __restrict__ offt,
                unsigned* __restrict__ Af) {
  const int Kc = 216;
  int unit = blockIdx.x * 8 + (threadIdx.x >> 5);
  int lane = threadIdx.x & 31;
  if (unit >= 8 * 1024 * 9) return;
  int tap = unit % 9;
  int pix = (unit / 9) & 1023;
  int b   = unit / (9 * 1024);
  int h = pix >> 5, w = pix & 31;
  int ky = tap / 3 - 1, kx = tap % 3 - 1;

  const float* ob = offt + (size_t)(b * 18 + 2 * tap) * 1024 + pix;
  float dy = ob[0];
  float dx = ob[1024];
  float py = (float)h + (float)ky + dy;
  float px = (float)w + (float)kx + dx;
  float fy = floorf(py), fx = floorf(px);
  int   y0 = (int)fy,    x0 = (int)fx;
  float wy1 = py - fy,   wx1 = px - fx;

  float wgt[4]; int idx[4];
  {
    int   ys[2] = {y0, y0 + 1},       xs[2] = {x0, x0 + 1};
    float wys[2] = {1.0f - wy1, wy1}, wxs[2] = {1.0f - wx1, wx1};
    int q = 0;
#pragma unroll
    for (int iy = 0; iy < 2; ++iy)
#pragma unroll
      for (int ix = 0; ix < 2; ++ix, ++q) {
        int yy = ys[iy], xx = xs[ix];
        bool ok = (yy >= 0) && (yy < 32) && (xx >= 0) && (xx < 32);
        int cy = min(max(yy, 0), 31), cx = min(max(xx, 0), 31);
        idx[q] = cy * 32 + cx;
        wgt[q] = wys[iy] * wxs[ix] * (ok ? 1.0f : 0.0f);
      }
  }

  const float* xb = x + (size_t)b * 768 * 1024;
  int m = b * 1024 + pix;
  int mt = m >> 4, mloc = m & 15;

  for (int pr = lane; pr < 384; pr += 32) {   // 384 channel pairs
    int c = pr * 2;
    const float* xc0 = xb + (size_t)c * 1024;
    const float* xc1 = xc0 + 1024;
    float v0 = wgt[0] * xc0[idx[0]] + wgt[1] * xc0[idx[1]] +
               wgt[2] * xc0[idx[2]] + wgt[3] * xc0[idx[3]];
    float v1 = wgt[0] * xc1[idx[0]] + wgt[1] * xc1[idx[1]] +
               wgt[2] * xc1[idx[2]] + wgt[3] * xc1[idx[3]];
    int k0 = tap * 768 + c;
    int kc = k0 >> 5;
    int p  = (k0 & 31) >> 1;
    int half = (p >> 2) & 1;
    int j    = (p & 3) | ((p & 8) ? 4 : 0);
    int dl   = mloc + 16 * half;
    Af[(((size_t)mt * Kc + kc) * 32 + dl) * 8 + j] =
        (unsigned)f2bf(v0) | ((unsigned)f2bf(v1) << 16);
  }
}

// ---------------------------------------------------------------------------
// im2col (3x3, pad 1) fp32 NCHW -> fragment-order bf16 A
// ---------------------------------------------------------------------------
__global__ void k_im2col(const float* __restrict__ in, unsigned* __restrict__ Af,
                         int C, int Hin, int Win, int Hout, int Wout, int stride) {
  int Kc = (9 * C) >> 5;
  size_t total = (size_t)8 * Hout * Wout * 9 * C / 2;
  size_t gs = (size_t)gridDim.x * blockDim.x;
  for (size_t i = (size_t)blockIdx.x * blockDim.x + threadIdx.x; i < total; i += gs) {
    int    j    = (int)(i & 7);
    int    lane = (int)((i >> 3) & 31);
    size_t tile = i >> 8;
    int    kc   = (int)(tile % Kc);
    int    mt   = (int)(tile / Kc);
    int half = lane >> 4;
    int m    = mt * 16 + (lane & 15);
    int k0   = kc * 32 + ((j >= 4) ? 16 : 0) + half * 8 + (j & 3) * 2;
    int tap  = k0 / C;
    int c    = k0 - tap * C;
    int x  = m % Wout;
    int r2 = m / Wout;
    int y  = r2 % Hout;
    int b  = r2 / Hout;
    int yy = y * stride + tap / 3 - 1;
    int xx = x * stride + tap % 3 - 1;
    unsigned lo = 0, hi = 0;
    if (yy >= 0 && yy < Hin && xx >= 0 && xx < Win) {
      size_t base = ((size_t)(b * C + c) * Hin + yy) * Win + xx;
      lo = f2bf(in[base]);
      hi = f2bf(in[base + (size_t)Hin * Win]);
    }
    Af[i] = lo | (hi << 16);
  }
}

// ---------------------------------------------------------------------------
// WMMA GEMM, double-buffered async A staging.
// block = 256 (8 waves); block tile 64(M) x 128(N); K chunk 32, unrolled x2.
// grid.x = M/64, grid.y = O/128.  Hout/Wout passed as log2 shifts.
// ---------------------------------------------------------------------------
__global__ void __launch_bounds__(256)
k_gemm(const uint4* __restrict__ Af, const uint4* __restrict__ Bf,
       const float* __restrict__ bias, float* __restrict__ out,
       int Kc, int O, int wsh, int hsh, int relu) {
  __shared__ uint4 sA[2][256];              // double-buffered 4 A tiles (2 x 4KB)
  int t    = threadIdx.x;
  int wave = t >> 5;
  int lane = t & 31;
  int mt0 = blockIdx.x * 4;
  int Nt  = O >> 4;
  int nt  = blockIdx.y * 8 + wave;

  v8f acc[4] = {{}, {}, {}, {}};

  int tileSel = t >> 6;
  int q       = t & 63;
  const uint4* aSrc0 = Af + ((size_t)(mt0 + tileSel) * Kc) * 64 + q;
  const uint4* bSrc0 = Bf + (size_t)nt * 64 + lane * 2;
  unsigned lds0 = (unsigned)(size_t)(&sA[0][t]);
  unsigned lds1 = (unsigned)(size_t)(&sA[1][t]);

  // prologue: chunk 0 -> buffer 0
  async_b128(lds0, aSrc0);

#define GEMM_STEP(KCV, RBUF, WLDS, KNEXT)                                      \
  {                                                                            \
    async_b128(WLDS, aSrc0 + (size_t)(KNEXT) * 64);       /* overlap copy */   \
    const uint4* bS = bSrc0 + (size_t)(KCV) * Nt * 64;                         \
    uint4 b0 = bS[0];                                                          \
    uint4 b1 = bS[1];                                                          \
    __builtin_prefetch((const void*)(bSrc0 + (size_t)(KNEXT) * Nt * 64), 0, 1);\
    wait_async_1();                           /* chunk KCV landed */           \
    __syncthreads();                                                           \
    Frag fb;                                                                   \
    fb.u = v8u{b0.x, b0.y, b0.z, b0.w, b1.x, b1.y, b1.z, b1.w};                \
    Frag fa[4];                                                                \
    _Pragma("unroll")                                                          \
    for (int i = 0; i < 4; ++i) {                                              \
      uint4 a0 = sA[RBUF][i * 64 + lane * 2];                                  \
      uint4 a1 = sA[RBUF][i * 64 + lane * 2 + 1];                              \
      fa[i].u = v8u{a0.x, a0.y, a0.z, a0.w, a1.x, a1.y, a1.z, a1.w};           \
    }                                                                          \
    _Pragma("unroll")                                                          \
    for (int i = 0; i < 4; ++i)                                                \
      acc[i] = __builtin_amdgcn_wmma_f32_16x16x32_bf16(                        \
          false, fa[i].b, false, fb.b, (short)0, acc[i], false, false);        \
    __syncthreads();                                                           \
  }

  for (int kc = 0; kc < Kc; kc += 2) {
    int kn2 = (kc + 2 < Kc) ? (kc + 2) : (kc + 1);   // clamped (dummy at tail)
    GEMM_STEP(kc,     0, lds1, kc + 1);
    GEMM_STEP(kc + 1, 1, lds0, kn2);
  }
  wait_async_0();                                     // drain dummy tail copy
#undef GEMM_STEP

  // ---- epilogue: bias (+relu), scatter to NCHW fp32 (pow2 dims -> shifts) ----
  int half8 = (lane & 16) ? 8 : 0;
  int n  = nt * 16 + (lane & 15);
  float bs = bias[n];
  int wmask = (1 << wsh) - 1;
  int hmask = (1 << hsh) - 1;
#pragma unroll
  for (int i = 0; i < 4; ++i) {
#pragma unroll
    for (int j = 0; j < 8; ++j) {
      int m = (mt0 + i) * 16 + j + half8;
      float v = acc[i][j] + bs;
      if (relu) v = fmaxf(v, 0.0f);
      int x = m & wmask;
      int y = (m >> wsh) & hmask;
      int b = m >> (wsh + hsh);
      out[(((size_t)(b * O + n)) << (wsh + hsh)) + (y << wsh) + x] = v;
    }
  }
}

// ---------------------------------------------------------------------------
// Host launcher
// ---------------------------------------------------------------------------
extern "C" void kernel_launch(void* const* d_in, const int* in_sizes, int n_in,
                              void* d_out, int out_size, void* d_ws, size_t ws_size,
                              hipStream_t stream) {
  (void)in_sizes; (void)n_in; (void)out_size; (void)ws_size;
  const float* cnn   = (const float*)d_in[0];
  const float* vit   = (const float*)d_in[1];
  const float* w_off = (const float*)d_in[2];
  const float* b_off = (const float*)d_in[3];
  const float* w_def = (const float*)d_in[4];
  const float* b_def = (const float*)d_in[5];
  const float* w1    = (const float*)d_in[6];
  const float* b1    = (const float*)d_in[7];
  const float* w2    = (const float*)d_in[8];
  const float* b2    = (const float*)d_in[9];
  float* out = (float*)d_out;

  char* ws = (char*)d_ws;
  auto alloc = [&](size_t bytes) {
    char* p = ws;
    ws += (bytes + 255) & ~(size_t)255;
    return p;
  };
  float*    offt = (float*)alloc((size_t)8 * 18 * 1024 * 4);
  float*    dt   = (float*)alloc((size_t)8 * 768 * 1024 * 4);
  float*    ht   = (float*)alloc((size_t)8 * 512 * 256 * 4);
  unsigned* Bdef = (unsigned*)alloc((size_t)6912 * 768 * 2);
  unsigned* B1   = (unsigned*)alloc((size_t)6912 * 512 * 2);
  unsigned* B2   = (unsigned*)alloc((size_t)4608 * 768 * 2);
  unsigned* Abuf = (unsigned*)alloc((size_t)8192 * 6912 * 2);   // reused 3x

  // ---- weight repack into fragment-order bf16 ----
  k_pack_w<<<(9 * 768 * 768 / 2 + 255) / 256, 256, 0, stream>>>(w_def, Bdef, 768, 768);
  k_pack_w<<<(9 * 768 * 512 / 2 + 255) / 256, 256, 0, stream>>>(w1, B1, 512, 768);
  k_pack_w<<<(9 * 512 * 768 / 2 + 255) / 256, 256, 0, stream>>>(w2, B2, 768, 512);

  // ---- offset conv ----
  k_offset_conv<<<8 * 18, 512, 0, stream>>>(vit, w_off, b_off, offt);

  // ---- deform sampling -> A_def (fragment order) ----
  k_deform_sample<<<(8 * 1024 * 9) / 8, 256, 0, stream>>>(cnn, offt, Abuf);

  // ---- deform GEMM: (8192 x 6912) x (6912 x 768) -> d (8,768,32,32) ----
  k_gemm<<<dim3(8192 / 64, 768 / 128), 256, 0, stream>>>(
      (const uint4*)Abuf, (const uint4*)Bdef, b_def, dt, 6912 / 32, 768, 5, 5, 0);

  // ---- conv1 (stride 2) + relu -> h (8,512,16,16) ----
  {
    size_t tot = (size_t)2048 * 6912 / 2;
    k_im2col<<<(int)((tot + 255) / 256), 256, 0, stream>>>(dt, Abuf, 768, 32, 32, 16, 16, 2);
  }
  k_gemm<<<dim3(2048 / 64, 512 / 128), 256, 0, stream>>>(
      (const uint4*)Abuf, (const uint4*)B1, b1, ht, 6912 / 32, 512, 4, 4, 1);

  // ---- conv2 (stride 1) -> out (8,768,16,16) ----
  {
    size_t tot = (size_t)2048 * 4608 / 2;
    k_im2col<<<(int)((tot + 255) / 256), 256, 0, stream>>>(ht, Abuf, 512, 16, 16, 16, 16, 1);
  }
  k_gemm<<<dim3(2048 / 64, 768 / 128), 256, 0, stream>>>(
      (const uint4*)Abuf, (const uint4*)B2, b2, out, 4608 / 32, 768, 4, 4, 0);
}
